// RCD_15152644620344
// MI455X (gfx1250) — matrix operand order repaired and checked by reading, not compiled, not run on hardware
//
#include <hip/hip_runtime.h>

typedef __attribute__((ext_vector_type(16))) _Float16 v16h;
typedef __attribute__((ext_vector_type(8)))  _Float16 v8h;
typedef __attribute__((ext_vector_type(8)))  float    v8f;

#define CDIM 512

// ---------- small helpers ----------
__device__ __forceinline__ unsigned f2ord(float f) {
    unsigned u = __float_as_uint(f);
    return (u & 0x80000000u) ? ~u : (u | 0x80000000u);
}
__device__ __forceinline__ float ord2f(unsigned u) {
    return (u & 0x80000000u) ? __uint_as_float(u & 0x7fffffffu) : __uint_as_float(~u);
}
__device__ __forceinline__ float sigmoidf(float x) { return 1.f / (1.f + __expf(-x)); }

__global__ void fill_f32(float* p, float v, int n) {
    int i = blockIdx.x * 256 + threadIdx.x;
    if (i < n) p[i] = v;
}
__global__ void fill_u32(unsigned* p, unsigned v, int n) {
    int i = blockIdx.x * 256 + threadIdx.x;
    if (i < n) p[i] = v;
}
__global__ void conv_f32_f16(const float* __restrict__ in, _Float16* __restrict__ out, int n) {
    int i = blockIdx.x * 256 + threadIdx.x;
    if (i < n) out[i] = (_Float16)in[i];
}
// Wt[n*K + k] = W[k*N + n]  (f32 -> f16, column-major stage for WMMA B operand)
__global__ void transp_f32_f16(const float* __restrict__ W, _Float16* __restrict__ Wt, int K, int N) {
    int n = blockIdx.x;
    for (int k = threadIdx.x; k < K; k += blockDim.x)
        Wt[(size_t)n * K + k] = (_Float16)W[(size_t)k * N + n];
}
__global__ void add_kernel(const float* a, const float* b, float* o, int n) {
    int i = blockIdx.x * 256 + threadIdx.x;
    if (i < n) o[i] = a[i] + b[i];
}
__global__ void gather_rows_f16(const float* __restrict__ src, const int* __restrict__ idx,
                                _Float16* __restrict__ dst) {
    int b = blockIdx.x;
    const float* r = src + (size_t)idx[b] * CDIM;
    for (int c = threadIdx.x; c < CDIM; c += blockDim.x)
        dst[(size_t)b * CDIM + c] = (_Float16)r[c];
}

// ---------- WMMA GEMM with async-to-LDS double-buffered staging ----------
//  C[MxN] = A[MxK] * B[KxN], B passed transposed (Bt[n][k]).
//  block = 256 threads = 8 waves (wave32); block tile 128(M) x 64(N); wave tile 32x32.
//  Per K-step (32): A tile 128x32 f16 (8KB) + B tile 64x32 f16 (4KB) staged into LDS
//  with GLOBAL_LOAD_ASYNC_TO_LDS_B128 (ASYNCcnt), double-buffered (24KB LDS).
__global__ void __launch_bounds__(256)
wmma_gemm_kernel(const _Float16* __restrict__ A, const _Float16* __restrict__ Bt,
                 float* __restrict__ C, int M, int N, int K)
{
    __shared__ alignas(32) char smem[24576];   // A: 2 x 8192 @0, B: 2 x 4096 @16384
    const int tid  = threadIdx.x;
    const int lane = tid & 31;
    const int wave = tid >> 5;
    const int m_blk = blockIdx.x * 128;
    const int n_blk = blockIdx.y * 64;
    const int m0 = (wave >> 1) * 32;   // wave tile origin within block tile
    const int n0 = (wave & 1) * 32;
    const int lr = lane & 15;
    const int hi = lane >> 4;          // lane group 0/1
    const int koffA = hi * 16;         // A frag: K = j + 16*hi (contiguous)
    const int koffB = hi * 16;         // B frag: K = j + 16*hi (contiguous)

    const unsigned ldsBase = (unsigned)(size_t)(&smem[0]);  // low 32 bits = LDS offset
    const unsigned ldsA = ldsBase;
    const unsigned ldsB = ldsBase + 16384;

    auto stage = [&](int step, int buf) {
        const int kb = step * 32;
        // A tile: 128 rows x 64B = 512 16B chunks, 2 per thread
#pragma unroll
        for (int c = 0; c < 2; ++c) {
            int chunk = tid + c * 256;
            int row = chunk >> 2, seg = chunk & 3;
            int grow = m_blk + row; if (grow >= M) grow = M - 1;   // clamp, masked on store
            unsigned long long g =
                (unsigned long long)A + ((size_t)grow * K + kb) * 2 + seg * 16;
            unsigned l = ldsA + buf * 8192 + row * 64 + seg * 16;
            asm volatile("global_load_async_to_lds_b128 %0, %1, off"
                         :: "v"(l), "v"(g) : "memory");
        }
        // B tile: 64 cols x 64B = 256 16B chunks, 1 per thread
        {
            int n = tid >> 2, seg = tid & 3;
            unsigned long long g =
                (unsigned long long)Bt + ((size_t)(n_blk + n) * K + kb) * 2 + seg * 16;
            unsigned l = ldsB + buf * 4096 + tid * 16;
            asm volatile("global_load_async_to_lds_b128 %0, %1, off"
                         :: "v"(l), "v"(g) : "memory");
        }
    };

    v8f acc[2][2] = {};
    const int NK = K >> 5;   // K/32 steps
    stage(0, 0);
    for (int s = 0; s < NK; ++s) {
        const int buf = s & 1;
        if (s + 1 < NK) {
            stage(s + 1, buf ^ 1);
            // 3 async loads in flight for next stage; wait until this stage's 3 landed
            asm volatile("s_wait_asynccnt 0x3" ::: "memory");
        } else {
            asm volatile("s_wait_asynccnt 0x0" ::: "memory");
        }
        __syncthreads();   // publish LDS tile to all waves

        const _Float16* As = (const _Float16*)(smem + buf * 8192);
        const _Float16* Bs = (const _Float16*)(smem + 16384 + buf * 4096);
        v16h afrag[2], bfrag[2];
#pragma unroll
        for (int i = 0; i < 2; ++i) {
            const _Float16* ap = As + (m0 + i * 16 + lr) * 32 + koffA;
            union { v16h v; v8h h[2]; } u;
            u.h[0] = *(const v8h*)(ap);        // K = koffA .. +7
            u.h[1] = *(const v8h*)(ap + 8);    // K = koffA+8 .. +15
            afrag[i] = u.v;
            bfrag[i] = *(const v16h*)(Bs + (n0 + i * 16 + lr) * 32 + koffB);
        }
#pragma unroll
        for (int i = 0; i < 2; ++i)
#pragma unroll
            for (int j = 0; j < 2; ++j)
                acc[i][j] = __builtin_amdgcn_wmma_f32_16x16x32_f16(
                    false, afrag[i], false, bfrag[j], (short)0, acc[i][j], false, false);
        __syncthreads();   // all waves done reading buf before it is re-staged
    }

    if (m_blk + 128 <= M) {   // fast path: whole block tile in range
#pragma unroll
        for (int i = 0; i < 2; ++i)
#pragma unroll
            for (int v = 0; v < 8; ++v) {
                int row = m_blk + m0 + i * 16 + hi * 8 + v;
#pragma unroll
                for (int j = 0; j < 2; ++j)
                    C[(size_t)row * N + n_blk + n0 + j * 16 + lr] = acc[i][j][v];
            }
    } else {
#pragma unroll
        for (int i = 0; i < 2; ++i)
#pragma unroll
            for (int v = 0; v < 8; ++v) {
                int row = m_blk + m0 + i * 16 + hi * 8 + v;
                if (row < M) {
#pragma unroll
                    for (int j = 0; j < 2; ++j)
                        C[(size_t)row * N + n_blk + n0 + j * 16 + lr] = acc[i][j][v];
                }
            }
    }
}

// ---------- GAT edge pipeline ----------
__global__ void edge_score_kernel(const float* __restrict__ z, const int* __restrict__ src,
                                  const int* __restrict__ dst, const float* __restrict__ avec,
                                  float* __restrict__ e, int nE)
{
    int eid = blockIdx.x * 8 + (threadIdx.x >> 5);
    int lane = threadIdx.x & 31;
    if (eid >= nE) return;
    const float* zs = z + (size_t)src[eid] * CDIM;
    const float* zd = z + (size_t)dst[eid] * CDIM;
    float s = 0.f;
    for (int c = lane; c < CDIM; c += 32)
        s += zs[c] * avec[c] + zd[c] * avec[CDIM + c];
    for (int o = 16; o; o >>= 1) s += __shfl_xor(s, o, 32);
    if (lane == 0) e[eid] = s > 0.f ? s : 0.01f * s;
}
__global__ void seg_max_kernel(const float* __restrict__ e, const int* __restrict__ dst,
                               unsigned* __restrict__ m_ord, int nE, int dst_lo)
{
    int i = blockIdx.x * 256 + threadIdx.x;
    if (i >= nE) return;
    atomicMax(&m_ord[dst[i] - dst_lo], f2ord(e[i]));
}
__global__ void seg_expsum_kernel(float* __restrict__ e, const int* __restrict__ dst,
                                  const unsigned* __restrict__ m_ord, float* __restrict__ den,
                                  int nE, int dst_lo)
{
    int i = blockIdx.x * 256 + threadIdx.x;
    if (i >= nE) return;
    int d = dst[i] - dst_lo;
    float ex = __expf(e[i] - ord2f(m_ord[d]));
    e[i] = ex;
    atomicAdd(&den[d], ex);
}
__global__ void seg_scatter_kernel(const float* __restrict__ z, const float* __restrict__ exv,
                                   const float* __restrict__ den, const int* __restrict__ src,
                                   const int* __restrict__ dst, float* __restrict__ out, int dst_lo)
{
    int eid = blockIdx.x;
    int d = dst[eid] - dst_lo;
    float alpha = exv[eid] / den[d];
    const float* zs = z + (size_t)src[eid] * CDIM;
    float* o = out + (size_t)d * CDIM;
    for (int c = threadIdx.x; c < CDIM; c += blockDim.x)
        atomicAdd(&o[c], alpha * zs[c]);
}

// ---------- attention fusion ----------
__global__ void fuse3_kernel(const float* __restrict__ kn, const float* __restrict__ A1,
                             const float* __restrict__ A2, const float* __restrict__ A3,
                             const float* __restrict__ aw, const float* __restrict__ ab,
                             float* __restrict__ out, int n)
{
    int row = blockIdx.x * 8 + (threadIdx.x >> 5);
    int lane = threadIdx.x & 31;
    if (row >= n) return;
    const float* k  = kn + (size_t)row * CDIM;
    const float* a1 = A1 + (size_t)row * CDIM;
    const float* a2 = A2 + (size_t)row * CDIM;
    const float* a3 = A3 + (size_t)row * CDIM;
    float s1 = 0.f, s2 = 0.f, s3 = 0.f;
    for (int c = lane; c < CDIM; c += 32) {
        float kv = k[c];
        s1 += kv * aw[c]        + a1[c] * aw[512 + c];
        s2 += kv * aw[1024 + c] + a2[c] * aw[1536 + c];
        s3 += kv * aw[2048 + c] + a3[c] * aw[2560 + c];
    }
    for (int o = 16; o; o >>= 1) {
        s1 += __shfl_xor(s1, o, 32);
        s2 += __shfl_xor(s2, o, 32);
        s3 += __shfl_xor(s3, o, 32);
    }
    s1 += ab[0]; s2 += ab[1]; s3 += ab[2];
    float mx = fmaxf(s1, fmaxf(s2, s3));
    float e1 = __expf(s1 - mx), e2 = __expf(s2 - mx), e3 = __expf(s3 - mx);
    float inv = 1.f / (e1 + e2 + e3);
    e1 *= inv; e2 *= inv; e3 *= inv;
    float* o = out + (size_t)row * CDIM;
    for (int c = lane; c < CDIM; c += 32)
        o[c] = k[c] + e1 * a1[c] + e2 * a2[c] + e3 * a3[c];
}
__global__ void fuse2_kernel(const float* __restrict__ ex, const float* __restrict__ Bq,
                             const float* __restrict__ Cq, const float* __restrict__ aw,
                             const float* __restrict__ ab, float* __restrict__ out, int n)
{
    int row = blockIdx.x * 8 + (threadIdx.x >> 5);
    int lane = threadIdx.x & 31;
    if (row >= n) return;
    const float* e  = ex + (size_t)row * CDIM;
    const float* b1 = Bq + (size_t)row * CDIM;
    const float* b2 = Cq + (size_t)row * CDIM;
    float s1 = 0.f, s2 = 0.f;
    for (int c = lane; c < CDIM; c += 32) {
        float ev = e[c];
        s1 += ev * aw[c]        + b1[c] * aw[512 + c];
        s2 += ev * aw[1024 + c] + b2[c] * aw[1536 + c];
    }
    for (int o = 16; o; o >>= 1) {
        s1 += __shfl_xor(s1, o, 32);
        s2 += __shfl_xor(s2, o, 32);
    }
    s1 += ab[0]; s2 += ab[1];
    float mx = fmaxf(s1, s2);
    float e1 = __expf(s1 - mx), e2 = __expf(s2 - mx);
    float inv = 1.f / (e1 + e2);
    e1 *= inv; e2 *= inv;
    float* o = out + (size_t)row * CDIM;
    for (int c = lane; c < CDIM; c += 32)
        o[c] = e[c] + e1 * b1[c] + e2 * b2[c];
}

// ---------- final prediction ----------
__global__ void __launch_bounds__(256)
predict_kernel(const float* __restrict__ P1, const float* __restrict__ K1,
               const float* __restrict__ P2, const float* __restrict__ K2,
               const float* __restrict__ W3, const float* __restrict__ b3,
               const float* __restrict__ q_table, const int* __restrict__ qid,
               float* __restrict__ out)
{
    __shared__ float sSum[8], sCnt[8];
    int b = blockIdx.x;
    int wave = threadIdx.x >> 5, lane = threadIdx.x & 31;
    const float* qrow = q_table + (size_t)qid[b] * CDIM;
    const float* p1 = P1 + (size_t)b * CDIM;
    const float* p2 = P2 + (size_t)b * CDIM;
    float bias = b3[0];
    float accS = 0.f, accC = 0.f;
    for (int k = wave; k < CDIM; k += 8) {
        const float* k1 = K1 + (size_t)k * CDIM;
        const float* k2 = K2 + (size_t)k * CDIM;
        float d = 0.f;
        for (int c = lane; c < CDIM; c += 32) {
            float pref = sigmoidf(p1[c] + k1[c]);
            float diff = sigmoidf(p2[c] + k2[c]);
            d += (pref - diff) * W3[c];
        }
        for (int o = 16; o; o >>= 1) d += __shfl_xor(d, o, 32);
        if (lane == 0) {
            float ov = sigmoidf(d + bias);
            float qv = qrow[k];
            accS += ov * qv; accC += qv;
        }
    }
    if (lane == 0) { sSum[wave] = accS; sCnt[wave] = accC; }
    __syncthreads();
    if (threadIdx.x == 0) {
        float s = 0.f, c = 0.f;
        for (int w = 0; w < 8; ++w) { s += sSum[w]; c += sCnt[w]; }
        out[b] = s / c;
    }
}

// ---------- host orchestration ----------
static inline int cdiv(int a, int b) { return (a + b - 1) / b; }

static void run_gat(const _Float16* A16, int M, const float* W, const float* avec,
                    const int* src, const int* dst, int nE, int dst_lo, int nDst,
                    _Float16* wT, float* z, float* escore, unsigned* m_ord, float* den,
                    float* out, hipStream_t s)
{
    transp_f32_f16<<<CDIM, 256, 0, s>>>(W, wT, CDIM, CDIM);
    dim3 g(cdiv(M, 128), CDIM / 64);
    wmma_gemm_kernel<<<g, 256, 0, s>>>(A16, wT, z, M, CDIM, CDIM);
    fill_f32<<<cdiv(nDst * CDIM, 256), 256, 0, s>>>(out, 0.f, nDst * CDIM);
    fill_u32<<<cdiv(nDst, 256), 256, 0, s>>>(m_ord, 0x007FFFFFu /* f2ord(-inf) */, nDst);
    fill_f32<<<cdiv(nDst, 256), 256, 0, s>>>(den, 0.f, nDst);
    edge_score_kernel<<<cdiv(nE, 8), 256, 0, s>>>(z, src, dst, avec, escore, nE);
    seg_max_kernel<<<cdiv(nE, 256), 256, 0, s>>>(escore, dst, m_ord, nE, dst_lo);
    seg_expsum_kernel<<<cdiv(nE, 256), 256, 0, s>>>(escore, dst, m_ord, den, nE, dst_lo);
    seg_scatter_kernel<<<nE, 256, 0, s>>>(z, escore, den, src, dst, out, dst_lo);
}

extern "C" void kernel_launch(void* const* d_in, const int* in_sizes, int n_in,
                              void* d_out, int out_size, void* d_ws, size_t ws_size,
                              hipStream_t stream)
{
    const int* user_id      = (const int*)d_in[0];
    const int* question_id  = (const int*)d_in[1];
    const float* q_table    = (const float*)d_in[2];
    const int* eDir = (const int*)d_in[3];
    const int* eUnd = (const int*)d_in[4];
    const int* eKE  = (const int*)d_in[5];
    const int* eEK  = (const int*)d_in[6];
    const int* eUE  = (const int*)d_in[7];
    const int* eEU  = (const int*)d_in[8];
    const float* user_emb     = (const float*)d_in[9];
    const float* question_emb = (const float*)d_in[10];
    const float* concept_emb  = (const float*)d_in[11];
    const float* gat_W  = (const float*)d_in[12];
    const float* gat_a  = (const float*)d_in[13];
    const float* attn_w = (const float*)d_in[14];
    const float* attn_b = (const float*)d_in[15];
    const float* W1 = (const float*)d_in[16];
    const float* W2 = (const float*)d_in[17];
    const float* W3 = (const float*)d_in[18];
    const float* b3 = (const float*)d_in[19];

    const int B = in_sizes[0];
    const int Q = in_sizes[10] / CDIM;
    const int U = in_sizes[9] / CDIM;
    const int nDir = in_sizes[3] / 2, nUnd = in_sizes[4] / 2;
    const int nKE = in_sizes[5] / 2, nEK = in_sizes[6] / 2;
    const int nUE = in_sizes[7] / 2, nEU = in_sizes[8] / 2;
    int maxE = nDir; if (nUnd > maxE) maxE = nUnd; if (nKE > maxE) maxE = nKE;
    if (nEK > maxE) maxE = nEK; if (nUE > maxE) maxE = nUE; if (nEU > maxE) maxE = nEU;
    const size_t C2 = (size_t)CDIM * CDIM;

    // workspace carve-out
    char* w = (char*)d_ws; size_t off = 0;
    auto alloc = [&](size_t bytes) -> void* {
        void* p = w + off; off = (off + bytes + 255) & ~(size_t)255; return p;
    };
    _Float16* eug16 = (_Float16*)alloc((size_t)(Q + U) * CDIM * 2);
    _Float16* ekg16 = (_Float16*)alloc((size_t)(Q + CDIM) * CDIM * 2);
    _Float16* wT    = (_Float16*)alloc(C2 * 2);
    float* z    = (float*)alloc((size_t)(Q + U) * CDIM * 4);
    float* knA  = (float*)alloc(C2 * 4);
    float* exA  = (float*)alloc((size_t)Q * CDIM * 4);
    float* stuA = (float*)alloc((size_t)U * CDIM * 4);
    float* knB  = (float*)alloc(C2 * 4);
    float* exB  = (float*)alloc((size_t)Q * CDIM * 4);
    float* stuB = (float*)alloc((size_t)U * CDIM * 4);
    float* kdir = (float*)alloc(C2 * 4);
    float* kund = (float*)alloc(C2 * 4);
    float* Dk   = (float*)alloc(C2 * 4);                    // kfe[Q:]
    float* efk  = (float*)alloc((size_t)Q * CDIM * 4);      // efk[:Q]
    float* ufe  = (float*)alloc((size_t)U * CDIM * 4);      // ufe[Q:]
    float* efu  = (float*)alloc((size_t)Q * CDIM * 4);      // efu[:Q]
    float* escore = (float*)alloc((size_t)maxE * 4);
    unsigned* m_ord = (unsigned*)alloc((size_t)U * 4);
    float* den = (float*)alloc((size_t)U * 4);
    _Float16* sb16 = (_Float16*)alloc((size_t)B * CDIM * 2);
    _Float16* eb16 = (_Float16*)alloc((size_t)B * CDIM * 2);
    _Float16* kn16p = (_Float16*)alloc(C2 * 2);
    float* P1 = (float*)alloc((size_t)B * CDIM * 4);
    float* P2 = (float*)alloc((size_t)B * CDIM * 4);
    float* K1 = (float*)alloc(C2 * 4);
    float* K2 = (float*)alloc(C2 * 4);
    (void)ws_size; (void)n_in; (void)out_size;

    const float* kn_in = concept_emb;
    const float* ex_in = question_emb;
    const float* stu_in = user_emb;
    float* knO[2] = { knA, knB };
    float* exO[2] = { exA, exB };
    float* stuO[2] = { stuA, stuB };

    for (int L = 0; L < 2; ++L) {
        // f16 staging: ekg = [ex; kn], eug = [ex; stu]
        conv_f32_f16<<<cdiv(Q * CDIM, 256), 256, 0, stream>>>(ex_in, ekg16, Q * CDIM);
        conv_f32_f16<<<cdiv(CDIM * CDIM, 256), 256, 0, stream>>>(kn_in, ekg16 + (size_t)Q * CDIM, CDIM * CDIM);
        conv_f32_f16<<<cdiv(Q * CDIM, 256), 256, 0, stream>>>(ex_in, eug16, Q * CDIM);
        conv_f32_f16<<<cdiv(U * CDIM, 256), 256, 0, stream>>>(stu_in, eug16 + (size_t)Q * CDIM, U * CDIM);
        const _Float16* kn16 = ekg16 + (size_t)Q * CDIM;

        const float* Wb = gat_W + (size_t)L * 6 * C2;
        const float* Ab = gat_a + (size_t)L * 6 * 1024;

        run_gat(kn16, CDIM, Wb + 0 * C2, Ab + 0 * 1024, eDir, eDir + nDir, nDir, 0, CDIM,
                wT, z, escore, m_ord, den, kdir, stream);
        run_gat(kn16, CDIM, Wb + 1 * C2, Ab + 1 * 1024, eUnd, eUnd + nUnd, nUnd, 0, CDIM,
                wT, z, escore, m_ord, den, kund, stream);
        run_gat(ekg16, Q + CDIM, Wb + 2 * C2, Ab + 2 * 1024, eKE, eKE + nKE, nKE, Q, CDIM,
                wT, z, escore, m_ord, den, Dk, stream);
        run_gat(ekg16, Q + CDIM, Wb + 3 * C2, Ab + 3 * 1024, eEK, eEK + nEK, nEK, 0, Q,
                wT, z, escore, m_ord, den, efk, stream);
        run_gat(eug16, Q + U, Wb + 4 * C2, Ab + 4 * 1024, eUE, eUE + nUE, nUE, Q, U,
                wT, z, escore, m_ord, den, ufe, stream);
        run_gat(eug16, Q + U, Wb + 5 * C2, Ab + 5 * 1024, eEU, eEU + nEU, nEU, 0, Q,
                wT, z, escore, m_ord, den, efu, stream);

        const float* awL = attn_w + (size_t)L * 5 * 1024;
        const float* abL = attn_b + (size_t)L * 5;
        fuse3_kernel<<<cdiv(CDIM, 8), 256, 0, stream>>>(kn_in, kdir, kund, Dk, awL, abL, knO[L], CDIM);
        fuse2_kernel<<<cdiv(Q, 8), 256, 0, stream>>>(ex_in, efk, efu, awL + 3 * 1024, abL + 3, exO[L], Q);
        add_kernel<<<cdiv(U * CDIM, 256), 256, 0, stream>>>(stu_in, ufe, stuO[L], U * CDIM);

        kn_in = knO[L]; ex_in = exO[L]; stu_in = stuO[L];
    }

    // prediction head
    gather_rows_f16<<<B, 256, 0, stream>>>(stuO[1], user_id, sb16);
    gather_rows_f16<<<B, 256, 0, stream>>>(exO[1], question_id, eb16);
    conv_f32_f16<<<cdiv(CDIM * CDIM, 256), 256, 0, stream>>>(knO[1], kn16p, CDIM * CDIM);

    dim3 gB(cdiv(B, 128), CDIM / 64), gC(cdiv(CDIM, 128), CDIM / 64);
    transp_f32_f16<<<CDIM, 256, 0, stream>>>(W1, wT, CDIM, CDIM);
    wmma_gemm_kernel<<<gB, 256, 0, stream>>>(sb16, wT, P1, B, CDIM, CDIM);
    transp_f32_f16<<<CDIM, 256, 0, stream>>>(W1 + C2, wT, CDIM, CDIM);
    wmma_gemm_kernel<<<gC, 256, 0, stream>>>(kn16p, wT, K1, CDIM, CDIM, CDIM);
    transp_f32_f16<<<CDIM, 256, 0, stream>>>(W2, wT, CDIM, CDIM);
    wmma_gemm_kernel<<<gB, 256, 0, stream>>>(eb16, wT, P2, B, CDIM, CDIM);
    transp_f32_f16<<<CDIM, 256, 0, stream>>>(W2 + C2, wT, CDIM, CDIM);
    wmma_gemm_kernel<<<gC, 256, 0, stream>>>(kn16p, wT, K2, CDIM, CDIM, CDIM);

    predict_kernel<<<B, 256, 0, stream>>>(P1, K1, P2, K2, W3, b3, q_table, question_id,
                                          (float*)d_out);
}
